// SDEPFCell_8864812499228
// MI455X (gfx1250) — matrix operand order repaired and checked by reading, not compiled, not run on hardware
//
#include <hip/hip_runtime.h>
#include <hip/hip_bf16.h>

// ---------------------------------------------------------------------------
// SDE particle-filter cell for MI455X (gfx1250, wave32, WMMA).
// B=256, K=256, H=128, I=64, O=64, N_UNFOLDS=6.
// ---------------------------------------------------------------------------

typedef __attribute__((ext_vector_type(16))) __bf16 v16bf;
typedef __attribute__((ext_vector_type(8)))  float  v8f;

#define Bc   256
#define Kc   256
#define Hc   128
#define Ic   64
#define Oc   64
#define NUc  6
#define BKc  (Bc * Kc)
#define ROWS_WG 128          // rows of s per workgroup (8 waves x 16 rows)
#define RSTRIDE 136          // bf16 row stride (=68 dwords): 16B aligned, 64-bank conflict-free

union FragU { uint4 q[2]; v16bf v; };

// load a v16bf fragment as two ds_load_b128 from a bf16 row, given dword offsets
static __device__ inline v16bf load_frag(const __bf16* rowbase, int d0, int d1) {
    const unsigned* p = (const unsigned*)rowbase;
    FragU u;
    u.q[0] = *(const uint4*)(p + d0);
    u.q[1] = *(const uint4*)(p + d1);
    return u.v;
}

// ---------------------------------------------------------------------------
// Kernel A: xWx[b][h] = x[b] @ Wx[:,h] + bias[h]   (tiny: 4 MFLOP)
// ---------------------------------------------------------------------------
__global__ __launch_bounds__(256) void xwx_kernel(
    const float* __restrict__ x, const float* __restrict__ Wx,
    const float* __restrict__ bias, float* __restrict__ xwx)
{
    int gid = blockIdx.x * 256 + threadIdx.x;   // 32768 = B*H
    int b = gid >> 7;
    int h = gid & 127;
    float acc = bias[h];
    #pragma unroll 8
    for (int i = 0; i < Ic; ++i)
        acc = fmaf(x[b * Ic + i], Wx[i * Hc + h], acc);
    xwx[gid] = acc;
}

// ---------------------------------------------------------------------------
// Kernel B: Euler-Maruyama propagation (6 unfolds of s@Wh) + output GEMM.
// 8 waves/block, 16 rows of s per wave, WMMA bf16 3-split (~fp32 accuracy).
// LDS planes (separate hi/lo, K-contiguous) -> fragments load as ds_load_b128.
// ---------------------------------------------------------------------------
__global__ __launch_bounds__(256) void prop_kernel(
    const float* __restrict__ particles,   // [BK,H]
    const float* __restrict__ noise,       // [BK,NU,H]
    const float* __restrict__ Wh,          // [H,H]
    const float* __restrict__ log_sigma,   // [H]
    const float* __restrict__ W_out,       // [H,O]
    const float* __restrict__ b_out,       // [O]
    const float* __restrict__ xwx,         // [B,H] (ws)
    float* __restrict__ out_particles,     // [BK,H] (d_out region 2)
    float* __restrict__ outputs_ws)        // [BK,O] (ws)
{
    extern __shared__ char smemB[];
    __bf16* wh_hi = (__bf16*)smemB;            // [128][RSTRIDE]  Wh^T hi (later W_out^T hi)
    __bf16* wh_lo = wh_hi + Hc * RSTRIDE;      // [128][RSTRIDE]  Wh^T lo
    __bf16* s_hi  = wh_lo + Hc * RSTRIDE;      // [128][RSTRIDE]  s hi
    __bf16* s_lo  = s_hi  + Hc * RSTRIDE;      // [128][RSTRIDE]  s lo

    const int tid  = threadIdx.x;
    const int wave = tid >> 5;
    const int lane = tid & 31;
    const int l16  = lane & 15;
    const int hw   = lane >> 4;             // half-wave index 0/1
    const int row0 = blockIdx.x * ROWS_WG;  // first global s row of this block
    const int b    = row0 / Kc;             // batch index (uniform per block)

    // stage Wh transposed: wh[n][k] = split(Wh[k][n])
    for (int i = tid; i < Hc * Hc; i += 256) {
        int k = i >> 7, n = i & 127;
        float v  = Wh[i];
        __bf16 h = (__bf16)v;
        wh_hi[n * RSTRIDE + k] = h;
        wh_lo[n * RSTRIDE + k] = (__bf16)(v - (float)h);
    }

    const float dt  = 1.0f / (float)NUc;
    const float sdt = sqrtf(dt);

    // per-lane hoisted per-h params (h = t*16 + l16 for n-tile t)
    float xwxv[8], gv[8];
    #pragma unroll
    for (int t = 0; t < 8; ++t) {
        int h = t * 16 + l16;
        xwxv[t] = xwx[b * Hc + h];
        float ls = log_sigma[h];
        float sp = (ls > 20.0f) ? ls : log1pf(expf(ls));  // softplus
        gv[t] = sp * sdt;
    }

    // load initial s in WMMA C layout: sreg[t][r] <-> (m = wave*16 + r + hw*8, h = t*16+l16)
    float sreg[8][8];
    #pragma unroll
    for (int t = 0; t < 8; ++t)
        #pragma unroll
        for (int r = 0; r < 8; ++r) {
            int m = wave * 16 + r + hw * 8;
            sreg[t][r] = particles[(size_t)(row0 + m) * Hc + t * 16 + l16];
        }

    __syncthreads();   // wh planes ready

    const int mrow = wave * 16 + l16;       // LDS row this lane reads for A fragments
    const __bf16* aHiRow = s_hi + mrow * RSTRIDE;
    const __bf16* aLoRow = s_lo + mrow * RSTRIDE;

    for (int u = 0; u < NUc; ++u) {
        // stage current s into LDS planes (per-wave region only, no barrier needed)
        #pragma unroll
        for (int t = 0; t < 8; ++t)
            #pragma unroll
            for (int r = 0; r < 8; ++r) {
                int m = wave * 16 + r + hw * 8;
                int h = t * 16 + l16;
                float v  = sreg[t][r];
                __bf16 hb = (__bf16)v;
                s_hi[m * RSTRIDE + h] = hb;
                s_lo[m * RSTRIDE + h] = (__bf16)(v - (float)hb);
            }

        // build A fragments (16x32 bf16 layout): dword j holds k=2j,2j+1 of its group
        v16bf aHi[4], aLo[4];
        #pragma unroll
        for (int c = 0; c < 4; ++c) {
            int d0 = c * 16 + hw * 4;       // VGPRs 0..3: k = c*32 + hw*8 + 0..7
            int d1 = d0 + 8;                // VGPRs 4..7: k = c*32 + 16 + hw*8 + 0..7
            aHi[c] = load_frag(aHiRow, d0, d1);
            aLo[c] = load_frag(aLoRow, d0, d1);
        }

        // 8 n-tiles of acc = s @ Wh
        #pragma unroll
        for (int t = 0; t < 8; ++t) {
            const __bf16* bHiRow = wh_hi + (t * 16 + l16) * RSTRIDE;
            const __bf16* bLoRow = wh_lo + (t * 16 + l16) * RSTRIDE;
            v8f acc = {};
            #pragma unroll
            for (int c = 0; c < 4; ++c) {
                int d0 = c * 16 + hw * 4;   // B: k = c*32 + hw*8 + 0..7, then c*32 + 16 + hw*8 + 0..7
                v16bf bHi = load_frag(bHiRow, d0, d0 + 8);
                v16bf bLo = load_frag(bLoRow, d0, d0 + 8);
                acc = __builtin_amdgcn_wmma_f32_16x16x32_bf16(false, aHi[c], false, bHi, (short)0, acc, false, false);
                acc = __builtin_amdgcn_wmma_f32_16x16x32_bf16(false, aHi[c], false, bLo, (short)0, acc, false, false);
                acc = __builtin_amdgcn_wmma_f32_16x16x32_bf16(false, aLo[c], false, bHi, (short)0, acc, false, false);
            }
            // elementwise EM update (A frags already captured old s -> safe to overwrite)
            #pragma unroll
            for (int r = 0; r < 8; ++r) {
                int m = wave * 16 + r + hw * 8;
                float z   = xwxv[t] + acc[r];
                float th  = tanhf(z);
                float so  = sreg[t][r];
                float eps = noise[((size_t)(row0 + m) * NUc + u) * Hc + t * 16 + l16];
                sreg[t][r] = so + (th - so) * dt + gv[t] * eps;
            }
        }
    }

    // write final (pre-resample) particles
    #pragma unroll
    for (int t = 0; t < 8; ++t)
        #pragma unroll
        for (int r = 0; r < 8; ++r) {
            int m = wave * 16 + r + hw * 8;
            out_particles[(size_t)(row0 + m) * Hc + t * 16 + l16] = sreg[t][r];
        }

    // stage final s for the output GEMM
    #pragma unroll
    for (int t = 0; t < 8; ++t)
        #pragma unroll
        for (int r = 0; r < 8; ++r) {
            int m = wave * 16 + r + hw * 8;
            int h = t * 16 + l16;
            float v  = sreg[t][r];
            __bf16 hb = (__bf16)v;
            s_hi[m * RSTRIDE + h] = hb;
            s_lo[m * RSTRIDE + h] = (__bf16)(v - (float)hb);
        }

    __syncthreads();   // everyone done with Wh -> recycle wh planes for W_out^T
    for (int i = tid; i < Hc * Oc; i += 256) {
        int k = i >> 6, o = i & 63;        // W_out is [H][O]
        float v  = W_out[i];
        __bf16 h = (__bf16)v;
        wh_hi[o * RSTRIDE + k] = h;
        wh_lo[o * RSTRIDE + k] = (__bf16)(v - (float)h);
    }
    __syncthreads();

    v16bf aHi[4], aLo[4];
    #pragma unroll
    for (int c = 0; c < 4; ++c) {
        int d0 = c * 16 + hw * 4;
        int d1 = d0 + 8;
        aHi[c] = load_frag(aHiRow, d0, d1);
        aLo[c] = load_frag(aLoRow, d0, d1);
    }

    #pragma unroll
    for (int t = 0; t < 4; ++t) {          // O = 64 -> 4 n-tiles
        const __bf16* bHiRow = wh_hi + (t * 16 + l16) * RSTRIDE;
        const __bf16* bLoRow = wh_lo + (t * 16 + l16) * RSTRIDE;
        v8f acc = {};
        #pragma unroll
        for (int c = 0; c < 4; ++c) {
            int d0 = c * 16 + hw * 4;
            v16bf bHi = load_frag(bHiRow, d0, d0 + 8);
            v16bf bLo = load_frag(bLoRow, d0, d0 + 8);
            acc = __builtin_amdgcn_wmma_f32_16x16x32_bf16(false, aHi[c], false, bHi, (short)0, acc, false, false);
            acc = __builtin_amdgcn_wmma_f32_16x16x32_bf16(false, aHi[c], false, bLo, (short)0, acc, false, false);
            acc = __builtin_amdgcn_wmma_f32_16x16x32_bf16(false, aLo[c], false, bHi, (short)0, acc, false, false);
        }
        float bo = b_out[t * 16 + l16];
        #pragma unroll
        for (int r = 0; r < 8; ++r) {
            int m = wave * 16 + r + hw * 8;
            outputs_ws[(size_t)(row0 + m) * Oc + t * 16 + l16] = acc[r] + bo;
        }
    }
}

// ---------------------------------------------------------------------------
// Kernel C: soft resampling + weighted output, one block per batch b.
// ---------------------------------------------------------------------------
static __device__ inline float block_sum(float v, float* red, int t) {
    __syncthreads();
    red[t] = v;
    __syncthreads();
    #pragma unroll
    for (int s = 128; s > 0; s >>= 1) {
        if (t < s) red[t] += red[t + s];
        __syncthreads();
    }
    return red[0];
}
static __device__ inline float block_max(float v, float* red, int t) {
    __syncthreads();
    red[t] = v;
    __syncthreads();
    #pragma unroll
    for (int s = 128; s > 0; s >>= 1) {
        if (t < s) red[t] = fmaxf(red[t], red[t + s]);
        __syncthreads();
    }
    return red[0];
}

__global__ __launch_bounds__(256) void resample_kernel(
    const float* __restrict__ log_weights,   // [B,K]
    const float* __restrict__ resample_u,    // [B,K]
    const float* __restrict__ alpha_logit,   // [1]
    const float* __restrict__ outputs_ws,    // [BK,O]
    float* __restrict__ out_output,          // [B,O]
    float* __restrict__ particles_io,        // [BK,H] in/out (gather in place via LDS)
    float* __restrict__ out_lw)              // [B,K]
{
    __shared__ float lw_s[Kc], q_s[Kc], cdf_s[Kc], wgt_s[Kc], red_s[256];
    __shared__ int   idx_s[Kc];
    extern __shared__ float pbuf[];          // 256*128 floats = 128 KB

    const int t = threadIdx.x;               // == particle index k
    const int b = blockIdx.x;

    // normalize log weights
    float lw0 = log_weights[b * Kc + t];
    float M   = block_max(lw0, red_s, t);
    float S   = block_sum(expf(lw0 - M), red_s, t);
    float lw  = lw0 - M - logf(S);
    lw_s[t] = lw;

    float w    = expf(lw);
    float sw2  = block_sum(w * w, red_s, t);
    float ess  = 1.0f / sw2;
    bool  need = ess < 0.5f * (float)Kc;

    float al    = alpha_logit[0];
    float alpha = 1.0f / (1.0f + expf(-al));
    float q     = alpha * w + (1.0f - alpha) / (float)Kc;
    q_s[t]   = q;
    cdf_s[t] = q;
    __syncthreads();

    // inclusive scan (Hillis-Steele)
    for (int off = 1; off < Kc; off <<= 1) {
        float add = (t >= off) ? cdf_s[t - off] : 0.0f;
        __syncthreads();
        cdf_s[t] += add;
        __syncthreads();
    }

    // inverse-CDF sampling: searchsorted(cdf, u) side='left'
    float uu = resample_u[b * Kc + t];
    int lo = 0, hi = Kc;
    while (lo < hi) {
        int mid = (lo + hi) >> 1;
        if (cdf_s[mid] < uu) lo = mid + 1; else hi = mid;
    }
    int idx = (lo > Kc - 1) ? (Kc - 1) : lo;
    idx_s[t] = idx;

    float lwr  = lw_s[idx] - logf(q_s[idx]);
    float M2   = block_max(lwr, red_s, t);
    float S2   = block_sum(expf(lwr - M2), red_s, t);
    float lwrn = lwr - M2 - logf(S2);

    float flw = need ? lwrn : lw;
    out_lw[b * Kc + t] = flw;

    // output weights
    float M3 = block_max(flw, red_s, t);
    float S3 = block_sum(expf(flw - M3), red_s, t);
    wgt_s[t] = expf(flw - M3 - logf(S3));

    // stage this batch's particles into LDS, then gather-write in place
    const float4* psrc4 = (const float4*)(particles_io + (size_t)b * Kc * Hc);
    float4*       pl4   = (float4*)pbuf;
    for (int i = t; i < Kc * (Hc / 4); i += 256) pl4[i] = psrc4[i];
    __syncthreads();

    float4* pdst4 = (float4*)(particles_io + (size_t)b * Kc * Hc);
    int lane32 = t & 31, grp = t >> 5;       // 8 rows per pass, 32 float4 per row
    for (int base = 0; base < Kc; base += 8) {
        int j  = base + grp;
        int sj = need ? idx_s[j] : j;
        pdst4[j * (Hc / 4) + lane32] = pl4[sj * (Hc / 4) + lane32];
    }

    // weighted average of (un-gathered) outputs: out[b][o] = sum_k wgt[k]*outputs[b,k,o]
    int o = t & 63, kg = t >> 6;
    float acc = 0.0f;
    for (int k = kg; k < Kc; k += 4)
        acc = fmaf(wgt_s[k], outputs_ws[((size_t)b * Kc + k) * Oc + o], acc);
    red_s[t] = acc;
    __syncthreads();
    if (t < 64)
        out_output[b * Oc + t] = red_s[t] + red_s[t + 64] + red_s[t + 128] + red_s[t + 192];
}

// ---------------------------------------------------------------------------
// Launch
// ---------------------------------------------------------------------------
extern "C" void kernel_launch(void* const* d_in, const int* in_sizes, int n_in,
                              void* d_out, int out_size, void* d_ws, size_t ws_size,
                              hipStream_t stream)
{
    const float* x           = (const float*)d_in[0];
    const float* particles   = (const float*)d_in[1];
    const float* log_weights = (const float*)d_in[2];
    const float* noise       = (const float*)d_in[3];
    const float* resample_u  = (const float*)d_in[4];
    const float* Wx          = (const float*)d_in[5];
    const float* Wh          = (const float*)d_in[6];
    const float* bias        = (const float*)d_in[7];
    const float* log_sigma   = (const float*)d_in[8];
    const float* W_out       = (const float*)d_in[9];
    const float* b_out       = (const float*)d_in[10];
    const float* alpha_logit = (const float*)d_in[11];

    float* out_output    = (float*)d_out;                       // [B,O]
    float* out_particles = out_output + Bc * Oc;                // [BK,H]
    float* out_lw        = out_particles + (size_t)BKc * Hc;    // [B,K]

    float* ws_xwx     = (float*)d_ws;                           // [B,H]
    float* ws_outputs = ws_xwx + Bc * Hc;                       // [BK,O]

    xwx_kernel<<<(Bc * Hc) / 256, 256, 0, stream>>>(x, Wx, bias, ws_xwx);

    size_t smemB = (size_t)4 * Hc * RSTRIDE * sizeof(__bf16);   // 4 planes = 136 KB
    prop_kernel<<<BKc / ROWS_WG, 256, smemB, stream>>>(
        particles, noise, Wh, log_sigma, W_out, b_out, ws_xwx,
        out_particles, ws_outputs);

    size_t smemC = (size_t)Kc * Hc * sizeof(float);             // 128 KB
    resample_kernel<<<Bc, 256, smemC, stream>>>(
        log_weights, resample_u, alpha_logit, ws_outputs,
        out_output, out_particles, out_lw);
}
